// GAT_37469294691200
// MI455X (gfx1250) — matrix operand order repaired and checked by reading, not compiled, not run on hardware
//
#include <hip/hip_runtime.h>
#include <hip/hip_bf16.h>
#include <math.h>

typedef __bf16 bf16_t;
typedef __attribute__((ext_vector_type(16))) __bf16 v16bf;
typedef __attribute__((ext_vector_type(8)))  float  v8f;

union Frag16 { v16bf v; unsigned u[8]; };

// ---------------------------------------------------------------- utilities
__global__ void k_fill(float* __restrict__ p, float v, long n) {
  long i = (long)blockIdx.x * blockDim.x + threadIdx.x;
  if (i < n) p[i] = v;
}

// fp32 [M,K] -> bf16 [M,KPad], zero-padding K
__global__ void k_cvt_act(const float* __restrict__ in, bf16_t* __restrict__ out,
                          int M, int K, int KPad) {
  long i = (long)blockIdx.x * blockDim.x + threadIdx.x;
  long total = (long)M * KPad;
  if (i >= total) return;
  int  k = (int)(i % KPad);
  long m = i / KPad;
  out[i] = (k < K) ? (bf16_t)in[m * (long)K + k] : (bf16_t)0.0f;
}

// fp32 W [K,Nout] -> bf16 W^T [Nout,KPad], zero-padding K
__global__ void k_cvt_wt(const float* __restrict__ W, bf16_t* __restrict__ out,
                         int K, int KPad, int Nout) {
  long i = (long)blockIdx.x * blockDim.x + threadIdx.x;
  long total = (long)Nout * KPad;
  if (i >= total) return;
  int k = (int)(i % KPad);
  int n = (int)(i / KPad);
  out[i] = (k < K) ? (bf16_t)W[(long)k * Nout + n] : (bf16_t)0.0f;
}

// ---------------------------------------------------------------- WMMA GEMM
// C[M,Nout] = A[M,KPad](bf16) * BT[Nout,KPad]^T(bf16) + bias[Nout]
// One wave computes a 32x32 block of C = 2x2 tiles of 16x16.
// Per K-step: 2 A-frags + 2 B-frags (8x b128) feed 4 WMMAs (2:1 load:wmma).
// KPad % 32 == 0, M % 16 == 0, Nout % 16 == 0.
__global__ void k_gemm_bf16(const bf16_t* __restrict__ A,
                            const bf16_t* __restrict__ BT,
                            const float*  __restrict__ bias,
                            float* __restrict__ C,
                            int M, int KPad, int Nout) {
  int tilesM = M >> 4, tilesN = Nout >> 4;
  int bN = (tilesN + 1) >> 1;
  int bM = (tilesM + 1) >> 1;
  int wave = blockIdx.x * (blockDim.x >> 5) + (threadIdx.x >> 5);
  int bm = wave / bN;
  int bn = wave - bm * bN;
  if (bm >= bM) return;
  int tm0 = bm * 2, tm1 = min(tm0 + 1, tilesM - 1);
  int tn0 = bn * 2, tn1 = min(tn0 + 1, tilesN - 1);

  int lane = threadIdx.x & 31;
  int l    = lane & 15;
  int hi   = lane >> 4;
  const bf16_t* a0 = A  + (long)(tm0 * 16 + l) * KPad;
  const bf16_t* a1 = A  + (long)(tm1 * 16 + l) * KPad;
  const bf16_t* b0 = BT + (long)(tn0 * 16 + l) * KPad;
  const bf16_t* b1 = BT + (long)(tn1 * 16 + l) * KPad;

  v8f c00 = {}, c01 = {}, c10 = {}, c11 = {};
  for (int k0 = 0; k0 < KPad; k0 += 32) {
    Frag16 fa0, fa1, fb0, fb1;
    // A 16x32 layout: elem i -> K = (i>>3)*16 + hi*8 + (i&7)
    int abase = k0 + hi * 8;
#pragma unroll
    for (int t = 0; t < 4; ++t) {
      fa0.u[t]     = *(const unsigned*)(a0 + abase + 2 * t);
      fa0.u[t + 4] = *(const unsigned*)(a0 + abase + 16 + 2 * t);
      fa1.u[t]     = *(const unsigned*)(a1 + abase + 2 * t);
      fa1.u[t + 4] = *(const unsigned*)(a1 + abase + 16 + 2 * t);
    }
    // B 32x16 layout: elem i -> K = (i>>3)*16 + hi*8 + (i&7)
    int bbase = k0 + hi * 8;
#pragma unroll
    for (int t = 0; t < 4; ++t) {
      fb0.u[t]     = *(const unsigned*)(b0 + bbase + 2 * t);
      fb0.u[t + 4] = *(const unsigned*)(b0 + bbase + 16 + 2 * t);
      fb1.u[t]     = *(const unsigned*)(b1 + bbase + 2 * t);
      fb1.u[t + 4] = *(const unsigned*)(b1 + bbase + 16 + 2 * t);
    }
    c00 = __builtin_amdgcn_wmma_f32_16x16x32_bf16(false, fa0.v, false, fb0.v,
                                                  (short)0, c00, false, false);
    c01 = __builtin_amdgcn_wmma_f32_16x16x32_bf16(false, fa0.v, false, fb1.v,
                                                  (short)0, c01, false, false);
    c10 = __builtin_amdgcn_wmma_f32_16x16x32_bf16(false, fa1.v, false, fb0.v,
                                                  (short)0, c10, false, false);
    c11 = __builtin_amdgcn_wmma_f32_16x16x32_bf16(false, fa1.v, false, fb1.v,
                                                  (short)0, c11, false, false);
  }

  // C/D layout: VGPR r -> row = hi*8 + r, col = lane&15 (within tile)
  int col0 = tn0 * 16 + l, col1 = tn1 * 16 + l;
  float bv0 = bias[col0], bv1 = bias[col1];
#pragma unroll
  for (int r = 0; r < 8; ++r) {
    int row0 = tm0 * 16 + hi * 8 + r;
    int row1 = tm1 * 16 + hi * 8 + r;
    C[(long)row0 * Nout + col0] = c00[r] + bv0;
    C[(long)row0 * Nout + col1] = c01[r] + bv1;
    C[(long)row1 * Nout + col0] = c10[r] + bv0;
    C[(long)row1 * Nout + col1] = c11[r] + bv1;
  }
}

// ---------------------------------------------------------------- edge passes
__device__ __forceinline__ void atomicMaxF(float* addr, float val) {
  if (val >= 0.0f) atomicMax((int*)addr, __float_as_int(val));
  else             atomicMin((unsigned int*)addr, __float_as_uint(val));
}

// pass 1: s[e,h] = att[h,:] . leaky_relu(xl[src]+xr[dst]) ; segment max into m[dst,h]
__global__ void k_edge_score(const float* __restrict__ xl, const float* __restrict__ xr,
                             const float* __restrict__ att, const int* __restrict__ ei,
                             int Erand, int Etot, int H, int C,
                             float* __restrict__ s, float* __restrict__ m) {
  long t = (long)blockIdx.x * blockDim.x + threadIdx.x;
  if (t >= (long)Etot * H) return;
  int e = (int)(t / H);
  int h = (int)(t - (long)e * H);
  int src, dst;
  if (e < Erand) { src = ei[e]; dst = ei[Erand + e]; }
  else           { src = e - Erand; dst = src; }          // self loops
  const float* pl = xl + ((long)src * H + h) * C;
  const float* pr = xr + ((long)dst * H + h) * C;
  const float* pa = att + (long)h * C;
  float acc = 0.f;
  for (int c = 0; c < C; ++c) {
    float v = pl[c] + pr[c];
    v = (v > 0.f) ? v : 0.2f * v;                          // leaky_relu, slope 0.2
    acc += pa[c] * v;
  }
  s[t] = acc;
  atomicMaxF(&m[(long)dst * H + h], acc);
}

// pass 2: ex = exp(s - m[dst]); den[dst,h] += ex   (ex overwrites s in place)
__global__ void k_edge_exp(const int* __restrict__ ei, int Erand, int Etot, int H,
                           const float* __restrict__ m, float* __restrict__ s,
                           float* __restrict__ den) {
  long t = (long)blockIdx.x * blockDim.x + threadIdx.x;
  if (t >= (long)Etot * H) return;
  int e = (int)(t / H);
  int h = (int)(t - (long)e * H);
  int dst = (e < Erand) ? ei[Erand + e] : (e - Erand);
  float ex = __expf(s[t] - m[(long)dst * H + h]);
  s[t] = ex;
  atomicAdd(&den[(long)dst * H + h], ex);
}

// pass 3: out[dst,h,:] += (ex/den[dst,h]) * xl[src,h,:]
__global__ void k_edge_scatter(const int* __restrict__ ei, int Erand, int Etot, int H, int C,
                               const float* __restrict__ ex, const float* __restrict__ den,
                               const float* __restrict__ xl, float* __restrict__ out) {
  long t = (long)blockIdx.x * blockDim.x + threadIdx.x;
  if (t >= (long)Etot * H) return;
  int e = (int)(t / H);
  int h = (int)(t - (long)e * H);
  int src, dst;
  if (e < Erand) { src = ei[e]; dst = ei[Erand + e]; }
  else           { src = e - Erand; dst = src; }
  float alpha = ex[t] / den[(long)dst * H + h];
  const float* pl = xl + ((long)src * H + h) * C;
  float*       po = out + ((long)dst * H + h) * C;
  for (int c = 0; c < C; ++c) atomicAdd(&po[c], alpha * pl[c]);
}

__global__ void k_finalize(float* __restrict__ out, const float* __restrict__ bias,
                           long total, int D, int relu) {
  long i = (long)blockIdx.x * blockDim.x + threadIdx.x;
  if (i >= total) return;
  int d = (int)(i % D);
  float v = out[i] + bias[d];
  if (relu) v = fmaxf(v, 0.f);
  out[i] = v;
}

// row softmax, one wave32 per row; D in {64,128}
__global__ void k_row_softmax(const float* __restrict__ in, float* __restrict__ out0,
                              float* __restrict__ out1, int N, int D) {
  int row = blockIdx.x * (blockDim.x >> 5) + (threadIdx.x >> 5);
  if (row >= N) return;
  int lane = threadIdx.x & 31;
  const float* p = in + (long)row * D;
  float vals[4];
  int cnt = 0;
  float mx = -INFINITY;
  for (int d = lane; d < D; d += 32) { float v = p[d]; vals[cnt++] = v; mx = fmaxf(mx, v); }
#pragma unroll
  for (int off = 16; off > 0; off >>= 1) mx = fmaxf(mx, __shfl_xor(mx, off, 32));
  float sum = 0.f;
  for (int i = 0; i < cnt; ++i) { vals[i] = __expf(vals[i] - mx); sum += vals[i]; }
#pragma unroll
  for (int off = 16; off > 0; off >>= 1) sum += __shfl_xor(sum, off, 32);
  float inv = 1.0f / sum;
  cnt = 0;
  for (int d = lane; d < D; d += 32) {
    float v = vals[cnt++] * inv;
    out0[(long)row * D + d] = v;
    if (out1) out1[(long)row * D + d] = v;
  }
}

// ---------------------------------------------------------------- launcher
struct LCfg { int din, H, C; };
static const LCfg g_cfg[10] = {
  {128, 16, 16}, {256, 25, 16}, {400, 16, 16}, {256, 1, 64}, {64, 1, 64},
  {64, 16, 64}, {1024, 25, 16}, {400, 16, 16}, {256, 1, 16}, {16, 1, 128}
};

static inline int gridFor(long n, int b) { return (int)((n + b - 1) / b); }

extern "C" void kernel_launch(void* const* d_in, const int* in_sizes, int n_in,
                              void* d_out, int out_size, void* d_ws, size_t ws_size,
                              hipStream_t stream) {
  const float* x  = (const float*)d_in[0];
  const int*   ei = (const int*)d_in[61];          // edge_index [2, Erand]
  int N     = in_sizes[0] / 128;
  int Erand = in_sizes[61] / 2;
  int Etot  = Erand + N;                            // + self loops
  const int MAXD = 1024, MAXH = 25;

  size_t off = 0;
  auto carve = [&](size_t bytes) -> char* {
    char* p = (char*)d_ws + off;
    off += (bytes + 255) & ~(size_t)255;
    return p;
  };
  float*  actA = (float*)carve((size_t)N * MAXD * 4);
  float*  actB = (float*)carve((size_t)N * MAXD * 4);
  float*  xl   = (float*)carve((size_t)N * MAXD * 4);
  float*  xr   = (float*)carve((size_t)N * MAXD * 4);
  float*  sbuf = (float*)carve((size_t)Etot * MAXH * 4);
  float*  mbuf = (float*)carve((size_t)N * MAXH * 4);
  float*  dbuf = (float*)carve((size_t)N * MAXH * 4);
  bf16_t* abf  = (bf16_t*)carve((size_t)N * MAXD * 2);
  bf16_t* wbf  = (bf16_t*)carve((size_t)1056 * 1024 * 2);
  (void)ws_size; (void)n_in; (void)out_size;

  float* outF = (float*)d_out;
  float* bufs[2] = {actA, actB};
  int pp = 0;
  const float* cur = x;

  for (int i = 0; i < 10; ++i) {
    int din = g_cfg[i].din, H = g_cfg[i].H, C = g_cfg[i].C;
    int hc = H * C;
    int KPad = (din + 31) & ~31;
    const float* Wl   = (const float*)d_in[1 + 6 * i + 0];
    const float* bl   = (const float*)d_in[1 + 6 * i + 1];
    const float* Wr   = (const float*)d_in[1 + 6 * i + 2];
    const float* br   = (const float*)d_in[1 + 6 * i + 3];
    const float* att  = (const float*)d_in[1 + 6 * i + 4];
    const float* bias = (const float*)d_in[1 + 6 * i + 5];

    // activations -> bf16 (K zero-padded)
    long na = (long)N * KPad;
    k_cvt_act<<<gridFor(na, 256), 256, 0, stream>>>(cur, abf, N, din, KPad);

    long nw = (long)hc * KPad;
    int bM = ((N >> 4) + 1) >> 1, bN = ((hc >> 4) + 1) >> 1;
    int waves  = bM * bN;
    int blocks = (waves + 7) / 8;                       // 8 waves / 256-thread block
    k_cvt_wt<<<gridFor(nw, 256), 256, 0, stream>>>(Wl, wbf, din, KPad, hc);
    k_gemm_bf16<<<blocks, 256, 0, stream>>>(abf, wbf, bl, xl, N, KPad, hc);
    k_cvt_wt<<<gridFor(nw, 256), 256, 0, stream>>>(Wr, wbf, din, KPad, hc);
    k_gemm_bf16<<<blocks, 256, 0, stream>>>(abf, wbf, br, xr, N, KPad, hc);

    float* out = bufs[pp];
    k_fill<<<gridFor((long)N * hc, 256), 256, 0, stream>>>(out, 0.f, (long)N * hc);
    k_fill<<<gridFor((long)N * H, 256), 256, 0, stream>>>(mbuf, -INFINITY, (long)N * H);
    k_fill<<<gridFor((long)N * H, 256), 256, 0, stream>>>(dbuf, 0.f, (long)N * H);

    long eh = (long)Etot * H;
    k_edge_score  <<<gridFor(eh, 256), 256, 0, stream>>>(xl, xr, att, ei, Erand, Etot, H, C, sbuf, mbuf);
    k_edge_exp    <<<gridFor(eh, 256), 256, 0, stream>>>(ei, Erand, Etot, H, mbuf, sbuf, dbuf);
    k_edge_scatter<<<gridFor(eh, 256), 256, 0, stream>>>(ei, Erand, Etot, H, C, sbuf, dbuf, xl, out);

    int relu = (i <= 3) || (i >= 5 && i <= 8);
    k_finalize<<<gridFor((long)N * hc, 256), 256, 0, stream>>>(out, bias, (long)N * hc, hc, relu);

    if (i == 4) {
      // h = softmax(layer4) : feeds layer5 and is 2nd output (after h1 = N*128 floats)
      k_row_softmax<<<gridFor(N, 8), 256, 0, stream>>>(out, bufs[pp ^ 1], outF + (long)N * 128, N, 64);
      cur = bufs[pp ^ 1];
    } else if (i == 9) {
      // h1 = softmax(layer9) : 1st output
      k_row_softmax<<<gridFor(N, 8), 256, 0, stream>>>(out, outF, nullptr, N, 128);
    } else {
      cur = out;
      pp ^= 1;
    }
  }
}
